// VladVQ_11879879544399
// MI455X (gfx1250) — matrix operand order repaired and checked by reading, not compiled, not run on hardware
//
#include <hip/hip_runtime.h>

#define D_DIM 512
#define K_CB  4096
#define N_TOK 16384
#define NBLK  1024          // N_TOK / 16 rows per block
#define ROWS_PREP (N_TOK + K_CB)

typedef __bf16 bf16_t;
typedef __attribute__((ext_vector_type(16))) __bf16 v16bf;
typedef __attribute__((ext_vector_type(8)))  float  v8f;
typedef __attribute__((ext_vector_type(4)))  unsigned int v4u;

union FragBF {
    v16bf v;
    v4u   q[2];
};

// -------- register-resident sorted top-8 (smallest distances) --------
struct Top8 { float d[8]; int i[8]; };

__device__ __forceinline__ void top8_init(Top8& t) {
#pragma unroll
    for (int k = 0; k < 8; ++k) { t.d[k] = 3.0e38f; t.i[k] = 0; }
}

__device__ __forceinline__ void top8_insert(Top8& t, float c, int ci) {
    if (c < t.d[7]) {
#pragma unroll
        for (int k = 0; k < 8; ++k) {
            bool lt = c < t.d[k];
            float td = t.d[k]; int ti = t.i[k];
            t.d[k] = lt ? c  : td;
            t.i[k] = lt ? ci : ti;
            c  = lt ? td : c;
            ci = lt ? ti : ci;
        }
    }
}

// -------- kernel 1: bf16 hi/lo split + row norms --------
__global__ __launch_bounds__(128)
void vq_prep(const float* __restrict__ x, const float* __restrict__ cb,
             bf16_t* __restrict__ x_hi, bf16_t* __restrict__ x_lo,
             bf16_t* __restrict__ cb_hi, bf16_t* __restrict__ cb_lo,
             float* __restrict__ x2, float* __restrict__ c2) {
    __shared__ float red[128];
    const int row = blockIdx.x;
    const float* src;
    bf16_t *hd, *ld;
    float* ndst;
    if (row < N_TOK) {
        src = x + (size_t)row * D_DIM;
        hd = x_hi + (size_t)row * D_DIM;
        ld = x_lo + (size_t)row * D_DIM;
        ndst = x2 + row;
    } else {
        const int r = row - N_TOK;
        src = cb + (size_t)r * D_DIM;
        hd = cb_hi + (size_t)r * D_DIM;
        ld = cb_lo + (size_t)r * D_DIM;
        ndst = c2 + r;
    }
    float s = 0.f;
    for (int e = threadIdx.x; e < D_DIM; e += 128) {
        float v = src[e];
        bf16_t h = (bf16_t)v;                 // RNE
        bf16_t l = (bf16_t)(v - (float)h);    // residual
        hd[e] = h;
        ld[e] = l;
        s += v * v;
    }
    red[threadIdx.x] = s;
    __syncthreads();
    for (int off = 64; off > 0; off >>= 1) {
        if ((int)threadIdx.x < off) red[threadIdx.x] += red[threadIdx.x + off];
        __syncthreads();
    }
    if (threadIdx.x == 0) *ndst = red[0];
}

// -------- kernel 2: fused distance-GEMM + top-8 + decode + loss --------
__global__ __launch_bounds__(256)
void vq_main(const float* __restrict__ x, const float* __restrict__ cb,
             const bf16_t* __restrict__ x_hi, const bf16_t* __restrict__ x_lo,
             const bf16_t* __restrict__ cb_hi, const bf16_t* __restrict__ cb_lo,
             const float* __restrict__ x2, const float* __restrict__ c2,
             float* __restrict__ partials, float* __restrict__ out) {
    __shared__ float dchunk[16][528];       // 16 rows x 512 cols, +16 pad (bank-safe)
    __shared__ float mergeD[16][16][8];
    __shared__ int   mergeI[16][16][8];
    __shared__ float selW[16][8];
    __shared__ int   selI[16][8];
    __shared__ float red[256];

    const int tid  = threadIdx.x;
    const int wav  = tid >> 5;        // wave id 0..7
    const int lane = tid & 31;
    const int half = lane >> 4;       // half-wave
    const int ln16 = lane & 15;       // A row / B col / C col within tile
    const int row_base = blockIdx.x * 16;

    // x row norms this lane contributes to: C/D VGPR r holds row m = r + 8*half
    float x2r[8];
#pragma unroll
    for (int r = 0; r < 8; ++r) x2r[r] = x2[row_base + half * 8 + r];

    // persistent per-thread top-8 (16 threads per token row, 256 cands each)
    Top8 sel;
    top8_init(sel);
    const int srow = tid >> 4;
    const int sj   = tid & 15;

    const bf16_t* aph = x_hi + (size_t)(row_base + ln16) * D_DIM;
    const bf16_t* apl = x_lo + (size_t)(row_base + ln16) * D_DIM;

    for (int g = 0; g < 8; ++g) {
        v8f acc[4] = {};
        const int colg = wav * 512 + g * 64;

        for (int ks = 0; ks < 16; ++ks) {
            // A fragment (16x32 bf16): lane m=ln16, k = ks*32 + 8*half + {0..7, 16..23}
            const int ka = ks * 32 + half * 8;
            FragBF ah, al;
            ah.q[0] = *(const v4u*)(aph + ka);
            ah.q[1] = *(const v4u*)(aph + ka + 16);
            al.q[0] = *(const v4u*)(apl + ka);
            al.q[1] = *(const v4u*)(apl + ka + 16);
            // B fragment (32x16 bf16): lane n=ln16, k = ks*32 + 8*half + {0..7, 16..23}
            const int kb = ks * 32 + half * 8;
#pragma unroll
            for (int t = 0; t < 4; ++t) {
                const bf16_t* bph = cb_hi + (size_t)(colg + t * 16 + ln16) * D_DIM + kb;
                const bf16_t* bpl = cb_lo + (size_t)(colg + t * 16 + ln16) * D_DIM + kb;
                FragBF bh, bl;
                bh.q[0] = *(const v4u*)(bph);
                bh.q[1] = *(const v4u*)(bph + 16);
                bl.q[0] = *(const v4u*)(bpl);
                bl.q[1] = *(const v4u*)(bpl + 16);
                // f32-class dot via hi*hi + lo*hi + hi*lo, f32 accumulate
                acc[t] = __builtin_amdgcn_wmma_f32_16x16x32_bf16(
                    false, ah.v, false, bh.v, (short)0, acc[t], false, false);
                acc[t] = __builtin_amdgcn_wmma_f32_16x16x32_bf16(
                    false, al.v, false, bh.v, (short)0, acc[t], false, false);
                acc[t] = __builtin_amdgcn_wmma_f32_16x16x32_bf16(
                    false, ah.v, false, bl.v, (short)0, acc[t], false, false);
            }
        }

        __syncthreads();   // previous selection pass done reading dchunk
#pragma unroll
        for (int t = 0; t < 4; ++t) {
            const float c2v = c2[colg + t * 16 + ln16];
#pragma unroll
            for (int r = 0; r < 8; ++r) {
                const float dist = x2r[r] - 2.0f * acc[t][r] + c2v;
                dchunk[half * 8 + r][wav * 64 + t * 16 + ln16] = dist;
            }
        }
        __syncthreads();

        // streaming top-8: thread (srow, sj) scans 32 of this chunk's 512 cols
        for (int s = 0; s < 32; ++s) {
            const int c = sj + (s << 4);
            const float dv = dchunk[srow][c];
            const int gcol = ((c >> 6) << 9) + g * 64 + (c & 63); // wave-span -> global col
            top8_insert(sel, dv, gcol);
        }
    }

    // merge 16 partial top-8 lists per row
#pragma unroll
    for (int i2 = 0; i2 < 8; ++i2) {
        mergeD[srow][sj][i2] = sel.d[i2];
        mergeI[srow][sj][i2] = sel.i[i2];
    }
    __syncthreads();

    if (tid < 16) {
        Top8 fin;
        top8_init(fin);
        for (int j2 = 0; j2 < 16; ++j2) {
#pragma unroll
            for (int i2 = 0; i2 < 8; ++i2)
                top8_insert(fin, mergeD[tid][j2][i2], mergeI[tid][j2][i2]);
        }
        // softmax denominator cancels: w_i = exp(d0 - d_i) / sum
        float wloc[8];
        float wsum = 0.f;
#pragma unroll
        for (int i2 = 0; i2 < 8; ++i2) {
            wloc[i2] = __expf(fin.d[0] - fin.d[i2]);
            wsum += wloc[i2];
        }
        const float inv = 1.0f / wsum;
#pragma unroll
        for (int i2 = 0; i2 < 8; ++i2) {
            selW[tid][i2] = wloc[i2] * inv;
            selI[tid][i2] = fin.i[i2];
        }
    }
    __syncthreads();

    // decode: q = sum_i w_i * codebook[idx_i] (f32 codebook, L2-hot), fused loss
    float w8[8];
    int   i8[8];
#pragma unroll
    for (int i2 = 0; i2 < 8; ++i2) { w8[i2] = selW[srow][i2]; i8[i2] = selI[srow][i2]; }
    const float* xr = x + (size_t)(row_base + srow) * D_DIM;
    float* qr = out + (size_t)(row_base + srow) * D_DIM;
    float lsum = 0.f;
    for (int dd = 0; dd < 32; ++dd) {
        const int dcol = sj + (dd << 4);          // 16 consecutive lanes -> coalesced
        float q = 0.f;
#pragma unroll
        for (int i2 = 0; i2 < 8; ++i2)
            q += w8[i2] * cb[(size_t)i8[i2] * D_DIM + dcol];
        const float xv = xr[dcol];
        const float df = q - xv;
        lsum += df * df;
        qr[dcol] = q;                              // STE forward value == quantized
    }

    red[tid] = lsum;
    __syncthreads();
    for (int off = 128; off > 0; off >>= 1) {
        if (tid < off) red[tid] += red[tid + off];
        __syncthreads();
    }
    if (tid == 0) partials[blockIdx.x] = red[0];
}

// -------- kernel 3: deterministic loss reduction --------
__global__ __launch_bounds__(256)
void vq_loss(const float* __restrict__ partials, float* __restrict__ loss_out) {
    __shared__ float red[256];
    float s = 0.f;
    for (int i = 0; i < NBLK / 256; ++i) s += partials[threadIdx.x + (i << 8)];
    red[threadIdx.x] = s;
    __syncthreads();
    for (int off = 128; off > 0; off >>= 1) {
        if ((int)threadIdx.x < off) red[threadIdx.x] += red[threadIdx.x + off];
        __syncthreads();
    }
    if (threadIdx.x == 0)
        loss_out[0] = red[0] * (1.25f / (float)(N_TOK * (size_t)D_DIM)); // (1 + 0.25) * mean
}

extern "C" void kernel_launch(void* const* d_in, const int* in_sizes, int n_in,
                              void* d_out, int out_size, void* d_ws, size_t ws_size,
                              hipStream_t stream) {
    const float* x  = (const float*)d_in[0];
    const float* cb = (const float*)d_in[1];

    char* ws = (char*)d_ws;
    size_t off = 0;
    bf16_t* x_hi  = (bf16_t*)(ws + off); off += (size_t)N_TOK * D_DIM * 2;
    bf16_t* x_lo  = (bf16_t*)(ws + off); off += (size_t)N_TOK * D_DIM * 2;
    bf16_t* cb_hi = (bf16_t*)(ws + off); off += (size_t)K_CB * D_DIM * 2;
    bf16_t* cb_lo = (bf16_t*)(ws + off); off += (size_t)K_CB * D_DIM * 2;
    float*  x2    = (float*)(ws + off);  off += (size_t)N_TOK * 4;
    float*  c2    = (float*)(ws + off);  off += (size_t)K_CB * 4;
    float*  parts = (float*)(ws + off);  off += (size_t)NBLK * 4;

    float* out = (float*)d_out;

    vq_prep<<<ROWS_PREP, 128, 0, stream>>>(x, cb, x_hi, x_lo, cb_hi, cb_lo, x2, c2);
    vq_main<<<NBLK, 256, 0, stream>>>(x, cb, x_hi, x_lo, cb_hi, cb_lo, x2, c2, parts, out);
    vq_loss<<<1, 256, 0, stream>>>(parts, out + (size_t)N_TOK * D_DIM);
}
